// SimplifiedCTM_34892314313367
// MI455X (gfx1250) — compile-verified, matrix operand order checked by reading, not run on hardware
//
#include <hip/hip_runtime.h>
#include <hip/hip_bf16.h>

// ---------------------------------------------------------------------------
// SimplifiedCTM forward on gfx1250 (MI455X), wave32 + WMMA bf16.
//  * Weights pre-swizzled once into packed WMMA-fragment layout (bf16 pairs in
//    u32), so GEMM B-tiles are staged with global_load_async_to_lds_b128.
//  * Each wave reads operands as contiguous 32B LDS fragments (2x ds_load_b128)
//    and issues 4 WMMAs per K-step (64x64 C tile per 128-thread workgroup).
//  * bnm,nmh->bnh einsum + LN + ReLU + nw2-dot fused; (B,N,128) never in HBM.
// ---------------------------------------------------------------------------

typedef __attribute__((ext_vector_type(16))) __bf16 v16bf;
typedef __attribute__((ext_vector_type(8)))  float  v8f;

static constexpr int Nn = 4096, Mm = 10, Tt = 10, Rr = 4, Pp = 1024, Aa = 64, Bb = 64;

// pack two floats into a bf16x2 dword (lo -> bits[15:0])
__device__ __forceinline__ unsigned int pack_bf16(float lo, float hi)
{
    unsigned short ul = __builtin_bit_cast(unsigned short, (__bf16)lo);
    unsigned short uh = __builtin_bit_cast(unsigned short, (__bf16)hi);
    return ((unsigned int)uh << 16) | (unsigned int)ul;
}

// ISA 7.12.2 16-bit operand layout. For even k: pair slot i and lane-half:
//   i = (k>=16 ? 4:0) + (k%8)/2 ,  h = (k>>3)&1 ,  lane = (row|col) + 16*h
__device__ __forceinline__ int frag_i(int k)  { return ((k & 16) >> 2) + ((k & 7) >> 1); }
__device__ __forceinline__ int frag_h(int k)  { return (k >> 3) & 1; }

// ---------------------------------------------------------------------------
// Pre-swizzle a (K x N) f32 weight matrix into tiled fragment layout:
//   out[((tk*Nt64 + tn)*4 + nt)*256 + lane*8 + p]  (bf16x2), zero padded.
// ---------------------------------------------------------------------------
__global__ void swizzle_weight(const float* __restrict__ W, unsigned int* __restrict__ out,
                               int Kdim, int Ndim, int Kt, int Nt64)
{
    int idx = blockIdx.x * blockDim.x + threadIdx.x;
    const int total = Kt * Nt64 * 1024;
    if (idx >= total) return;
    const int p    = idx & 7;
    const int lane = (idx >> 3) & 31;
    const int nt   = (idx >> 8) & 3;
    const int tile = idx >> 10;
    const int tn   = tile % Nt64;
    const int tk   = tile / Nt64;
    const int cc = lane & 15, hh = lane >> 4;
    const int k  = tk * 32 + ((p & 4) << 2) + ((p & 3) << 1) + (hh << 3);
    const int n  = tn * 64 + nt * 16 + cc;
    const float lo = (k     < Kdim && n < Ndim) ? W[(size_t)k * Ndim + n] : 0.f;
    const float hi = (k + 1 < Kdim && n < Ndim) ? W[(size_t)(k + 1) * Ndim + n] : 0.f;
    out[idx] = pack_bf16(lo, hi);
}

// ---------------------------------------------------------------------------
// C(64,N) = A(64,K) @ B(K,N) + bias.  128 threads (4 waves) per 64x64 C tile.
// B comes pre-swizzled; its tile is staged by the async copy engine.
// ---------------------------------------------------------------------------
__global__ void gemm64_wmma(const float* __restrict__ A,
                            const unsigned int* __restrict__ Bsw,
                            const float* __restrict__ bias,
                            float* __restrict__ C,
                            int Kdim, int Ndim, int Nt64)
{
    __shared__ alignas(32) unsigned int Afrag[4][256];   // 4 M-tiles
    __shared__ alignas(32) unsigned int Bfrag[4][256];   // 4 N-tiles
    const int n0   = blockIdx.x * 64;
    const int tid  = threadIdx.x;
    const int lane = tid & 31;
    const int w    = tid >> 5;            // wave id == M-tile
    const int r    = lane & 15, h = lane >> 4;
    const int Kt   = (Kdim + 31) >> 5;

    v8f acc[4] = {{}, {}, {}, {}};

    for (int tk = 0; tk < Kt; ++tk) {
        const int k0 = tk * 32;

        // ---- B tile: 4KB contiguous async copy, global -> LDS, no VGPR data
        {
            const unsigned int* src =
                Bsw + ((size_t)tk * Nt64 + blockIdx.x) * 1024 + (size_t)tid * 8;
            unsigned lds = (unsigned)(unsigned long long)(const void*)(&Bfrag[0][0] + tid * 8);
            unsigned long long ga = (unsigned long long)src;
            asm volatile("global_load_async_to_lds_b128 %0, %1, off"
                         :: "v"(lds), "v"(ga) : "memory");
            asm volatile("global_load_async_to_lds_b128 %0, %1, off offset:16"
                         :: "v"(lds), "v"(ga) : "memory");
        }

        // ---- A tile (64x32): f32 -> packed bf16 fragments
        {
            const int row = tid >> 1;
            const int c0  = (tid & 1) * 16;
            const int mt  = row >> 4, rr = row & 15;
            float f[16];
            if (k0 + 32 <= Kdim) {
                const float4* src = (const float4*)(A + (size_t)row * Kdim + k0 + c0);
#pragma unroll
                for (int q = 0; q < 4; ++q) {
                    float4 v = src[q];
                    f[4 * q + 0] = v.x; f[4 * q + 1] = v.y;
                    f[4 * q + 2] = v.z; f[4 * q + 3] = v.w;
                }
            } else {
#pragma unroll
                for (int q = 0; q < 16; ++q) {
                    const int gk = k0 + c0 + q;
                    f[q] = (gk < Kdim) ? A[(size_t)row * Kdim + gk] : 0.f;
                }
            }
#pragma unroll
            for (int p2 = 0; p2 < 8; ++p2) {
                const int k = c0 + 2 * p2;
                Afrag[mt][(rr + (frag_h(k) << 4)) * 8 + frag_i(k)] =
                    pack_bf16(f[2 * p2], f[2 * p2 + 1]);
            }
            if (k0 + 32 < Kdim)
                __builtin_prefetch(A + (size_t)row * Kdim + k0 + 32 + c0, 0, 1);
        }

        asm volatile("s_wait_asynccnt 0x0" ::: "memory");
        __syncthreads();

        const v16bf a = *(const v16bf*)(&Afrag[w][0] + (size_t)lane * 8);
#pragma unroll
        for (int nt = 0; nt < 4; ++nt) {
            const v16bf b = *(const v16bf*)(&Bfrag[nt][0] + (size_t)lane * 8);
            acc[nt] = __builtin_amdgcn_wmma_f32_16x16x32_bf16(false, a, false, b,
                                                              (short)0, acc[nt],
                                                              false, false);
        }
        __syncthreads();
    }

    // ---- store: lane holds (M = w*16 + 8h + j, N = n0 + nt*16 + r)
#pragma unroll
    for (int nt = 0; nt < 4; ++nt) {
        const int gn = n0 + nt * 16 + r;
        if (gn < Ndim) {
            const float bv = bias[gn];
#pragma unroll
            for (int j = 0; j < 8; ++j) {
                const int gm = w * 16 + h * 8 + j;
                C[(size_t)gm * Ndim + gn] = acc[nt][j] + bv;
            }
        }
    }
}

// ---------------------------------------------------------------------------
// Fused neuron tick: per n, H = pre[:,n,:] (64x10pad32) @ nw1[n] (10x128)
// + nb1[n], then LN + ReLU + dot(nw2[n]) + nb2[n] -> post_new[:,n].
// 8 waves; each wave: 1 A fragment, 4 WMMAs.  H stays in LDS.
// ---------------------------------------------------------------------------
__global__ void neuron_tick(const float* __restrict__ pre,   // (B,N,M)
                            const float* __restrict__ nw1,   // (N,M,128)
                            const float* __restrict__ nb1,   // (N,128)
                            const float* __restrict__ ng1,
                            const float* __restrict__ nbe1,
                            const float* __restrict__ nw2,   // (N,128)
                            const float* __restrict__ nb2,   // (N,)
                            float* __restrict__ post_new)    // (B,N)
{
    const int n    = blockIdx.x;
    const int tid  = threadIdx.x;      // 256
    const int lane = tid & 31;
    const int w    = tid >> 5;         // 0..7

    __shared__ alignas(32) unsigned int Afrag[4][256];   // 4KB
    __shared__ alignas(32) unsigned int Bfrag[8][256];   // 8KB
    __shared__ float Hs[64 * 128];                       // 32KB

    // ---- stage A: pre[:, n, 0..9], K padded to 32
    {
        const int row = tid >> 2;            // 0..63
        const int c0  = (tid & 3) * 8;       // 0,8,16,24
        const int mt  = row >> 4, rr = row & 15;
        const float* src = pre + (size_t)row * Nn * Mm + (size_t)n * Mm;
        float f[8];
#pragma unroll
        for (int q = 0; q < 8; ++q) {
            const int k = c0 + q;
            f[q] = (k < Mm) ? src[k] : 0.f;
        }
#pragma unroll
        for (int p2 = 0; p2 < 4; ++p2) {
            const int k = c0 + 2 * p2;
            Afrag[mt][(rr + (frag_h(k) << 4)) * 8 + frag_i(k)] =
                pack_bf16(f[2 * p2], f[2 * p2 + 1]);
        }
    }
    // ---- stage B: nw1[n] (10x128), K padded to 32
    {
        const int kk = (tid >> 4) * 2;       // 0,2,...,30
        const int c0 = (tid & 15) * 8;       // 0..120
        const float* src = nw1 + (size_t)n * Mm * 128;
        float f0[8], f1[8];
#pragma unroll
        for (int q = 0; q < 8; ++q) {
            f0[q] = (kk     < Mm) ? src[(size_t)kk * 128 + c0 + q]       : 0.f;
            f1[q] = (kk + 1 < Mm) ? src[(size_t)(kk + 1) * 128 + c0 + q] : 0.f;
        }
        const int hh = frag_h(kk), ip = frag_i(kk);
#pragma unroll
        for (int q = 0; q < 8; ++q) {
            const int c = c0 + q;
            Bfrag[c >> 4][((c & 15) + (hh << 4)) * 8 + ip] = pack_bf16(f0[q], f1[q]);
        }
    }
    __syncthreads();

    // ---- compute: wave w -> M-tile w>>1, N-tiles (w&1)*4 .. +3
    {
        const int r = lane & 15, h = lane >> 4;
        const int tm = w >> 1, tn0 = (w & 1) * 4;
        const v16bf a = *(const v16bf*)(&Afrag[tm][0] + (size_t)lane * 8);
#pragma unroll
        for (int q = 0; q < 4; ++q) {
            const int nt = tn0 + q;
            const v16bf b = *(const v16bf*)(&Bfrag[nt][0] + (size_t)lane * 8);
            v8f acc = {};
            acc = __builtin_amdgcn_wmma_f32_16x16x32_bf16(false, a, false, b,
                                                          (short)0, acc, false, false);
            const float bv = nb1[(size_t)n * 128 + nt * 16 + r];
#pragma unroll
            for (int j = 0; j < 8; ++j)
                Hs[(tm * 16 + h * 8 + j) * 128 + nt * 16 + r] = acc[j] + bv;
        }
    }
    __syncthreads();

    // ---- LN + ReLU + dot(nw2) per row of 128; one wave per row
    for (int row = w; row < 64; row += 8) {
        float vals[4];
        float s = 0.f, ss = 0.f;
#pragma unroll
        for (int j = 0; j < 4; ++j) {
            float v = Hs[row * 128 + j * 32 + lane];
            vals[j] = v; s += v; ss += v * v;
        }
#pragma unroll
        for (int off = 16; off > 0; off >>= 1) {
            s  += __shfl_xor(s,  off, 32);
            ss += __shfl_xor(ss, off, 32);
        }
        const float mean = s * (1.f / 128.f);
        const float var  = ss * (1.f / 128.f) - mean * mean;
        const float inv  = rsqrtf(var + 1e-5f);
        float dot = 0.f;
#pragma unroll
        for (int j = 0; j < 4; ++j) {
            const int col = j * 32 + lane;
            float v = (vals[j] - mean) * inv * ng1[(size_t)n * 128 + col]
                      + nbe1[(size_t)n * 128 + col];
            v = fmaxf(v, 0.f);
            dot += v * nw2[(size_t)n * 128 + col];
        }
#pragma unroll
        for (int off = 16; off > 0; off >>= 1) dot += __shfl_xor(dot, off, 32);
        if (lane == 0) post_new[(size_t)row * Nn + n] = dot + nb2[n];
    }
}

// ---------------------------------------------------------------------------
// Per-row LayerNorm + ReLU, in place.  One wave per row, D <= 512.
// ---------------------------------------------------------------------------
__global__ void ln_relu_rows(float* __restrict__ buf,
                             const float* __restrict__ g,
                             const float* __restrict__ be,
                             int rows, int D)
{
    const int row = blockIdx.x;
    const int lane = threadIdx.x;
    float vals[16];
    float s = 0.f, ss = 0.f;
    int cnt = 0;
    for (int c = lane; c < D; c += 32) {
        float v = buf[(size_t)row * D + c];
        vals[cnt++] = v; s += v; ss += v * v;
    }
#pragma unroll
    for (int off = 16; off > 0; off >>= 1) {
        s  += __shfl_xor(s,  off, 32);
        ss += __shfl_xor(ss, off, 32);
    }
    const float mean = s / (float)D;
    const float var  = ss / (float)D - mean * mean;
    const float inv  = rsqrtf(var + 1e-5f);
    cnt = 0;
    for (int c = lane; c < D; c += 32) {
        float v = (vals[cnt++] - mean) * inv * g[c] + be[c];
        buf[(size_t)row * D + c] = fmaxf(v, 0.f);
    }
}

// ---------------------------------------------------------------------------
// Small helper kernels
// ---------------------------------------------------------------------------
__global__ void broadcast_init(const float* __restrict__ init, float* __restrict__ buf)
{
    const size_t total = (size_t)Bb * Nn * Mm;
    size_t idx = (size_t)blockIdx.x * blockDim.x + threadIdx.x;
    if (idx < total) buf[idx] = init[idx % ((size_t)Nn * Mm)];
}

__global__ void zero_int(int* p) { *p = 0; }

__global__ void build_syn_in(const float* __restrict__ enc,
                             const float* __restrict__ post,
                             float* __restrict__ syn_in)
{
    const int W = Rr + Nn;
    int idx = blockIdx.x * blockDim.x + threadIdx.x;
    if (idx >= Bb * W) return;
    const int b = idx / W, c = idx % W;
    syn_in[idx] = (c < Rr) ? enc[b * Rr + c]
                           : post[(size_t)b * Nn * Mm + (size_t)(c - Rr) * Mm + (Mm - 1)];
}

__global__ void shift_append(float* __restrict__ buf,          // (B,N,M)
                             const float* __restrict__ newcol) // (B,N)
{
    int idx = blockIdx.x * blockDim.x + threadIdx.x;
    if (idx >= Bb * Nn) return;
    float* p = buf + (size_t)idx * Mm;
#pragma unroll
    for (int m = 0; m < Mm - 1; ++m) p[m] = p[m + 1];
    p[Mm - 1] = newcol[idx];
}

__global__ void sync_kernel(const float* __restrict__ post,    // (B,N,M)
                            const int* __restrict__ pairs,     // (P,2)
                            const float* __restrict__ decay,   // (P,)
                            int t, float* __restrict__ syncv)  // (B,P)
{
    int idx = blockIdx.x * blockDim.x + threadIdx.x;
    if (idx >= Bb * Pp) return;
    const int b = idx / Pp, p = idx % Pp;
    const int i = pairs[p * 2], j = pairs[p * 2 + 1];
    const float* Zi = post + (size_t)b * Nn * Mm + (size_t)i * Mm;
    const float* Zj = post + (size_t)b * Nn * Mm + (size_t)j * Mm;
    const float d = decay[p];
    const int hlen = (t + 1 < Mm) ? t + 1 : Mm;
    float wdp = 0.f, norm = 0.f;
#pragma unroll
    for (int m = 0; m < Mm; ++m) {
        const float td = (float)(Mm - 1 - m);
        const float mask = (m >= Mm - hlen) ? 1.f : 0.f;
        const float rw = __expf(-d * td) * mask;
        wdp  += Zi[m] * rw * Zj[m];
        norm += rw;
    }
    syncv[idx] = wdp / (norm + 1e-8f);
}

__global__ void kv_kernel(const float* __restrict__ enc,
                          const float* __restrict__ kw, const float* __restrict__ kb,
                          const float* __restrict__ vw, const float* __restrict__ vb,
                          float* __restrict__ keys, float* __restrict__ values)
{
    int idx = blockIdx.x * blockDim.x + threadIdx.x;
    if (idx >= Bb * Aa) return;
    const int b = idx / Aa, a = idx % Aa;
    float kacc = kb[a], vacc = vb[a];
#pragma unroll
    for (int r = 0; r < Rr; ++r) {
        const float e = enc[b * Rr + r];
        kacc += e * kw[r * Aa + a];
        vacc += e * vw[r * Aa + a];
    }
    keys[idx] = kacc;
    values[idx] = vacc;
}

__global__ void build_comb(const float* __restrict__ values,
                           const float* __restrict__ syncv,
                           float* __restrict__ comb)
{
    const int W = Aa + Pp;
    int idx = blockIdx.x * blockDim.x + threadIdx.x;
    if (idx >= Bb * W) return;
    const int b = idx / W, c = idx % W;
    comb[idx] = (c < Aa) ? values[b * Aa + c] : syncv[b * Pp + (c - Aa)];
}

__global__ void finalize_kernel(const float* __restrict__ pred, // (B,10)
                                float* __restrict__ stored,     // d_out (B,10)
                                int* __restrict__ done_flag)
{
    __shared__ int done_prev;
    __shared__ int anyflag;
    const int b = threadIdx.x;    // 64 threads
    if (b == 0) { done_prev = *done_flag; anyflag = 0; }
    __syncthreads();
    float row[10];
    float mx = -1e30f;
#pragma unroll
    for (int c = 0; c < 10; ++c) { row[c] = pred[b * 10 + c]; mx = fmaxf(mx, row[c]); }
    float sum = 0.f;
#pragma unroll
    for (int c = 0; c < 10; ++c) sum += __expf(row[c] - mx);
    const float maxc = 1.f / sum;             // max of softmax = exp(0)/sum
    if (!done_prev)
#pragma unroll
        for (int c = 0; c < 10; ++c) stored[b * 10 + c] = row[c];
    if (maxc > 0.8f) atomicOr(&anyflag, 1);
    __syncthreads();
    if (b == 0 && anyflag) *done_flag = 1;
}

// ---------------------------------------------------------------------------
// Host orchestration
// ---------------------------------------------------------------------------
static inline int cdiv(int a, int b) { return (a + b - 1) / b; }

extern "C" void kernel_launch(void* const* d_in, const int* in_sizes, int n_in,
                              void* d_out, int out_size, void* d_ws, size_t ws_size,
                              hipStream_t stream)
{
    const float* x       = (const float*)d_in[0];
    const float* ie_w1   = (const float*)d_in[1];
    const float* ie_b1   = (const float*)d_in[2];
    const float* ie_g1   = (const float*)d_in[3];
    const float* ie_be1  = (const float*)d_in[4];
    const float* ie_w2   = (const float*)d_in[5];
    const float* ie_b2   = (const float*)d_in[6];
    const float* ie_g2   = (const float*)d_in[7];
    const float* ie_be2  = (const float*)d_in[8];
    const float* ie_w3   = (const float*)d_in[9];
    const float* ie_b3   = (const float*)d_in[10];
    const float* sy_w1   = (const float*)d_in[11];
    const float* sy_b1   = (const float*)d_in[12];
    const float* sy_g1   = (const float*)d_in[13];
    const float* sy_be1  = (const float*)d_in[14];
    const float* sy_w2   = (const float*)d_in[15];
    const float* sy_b2   = (const float*)d_in[16];
    const float* nw1     = (const float*)d_in[17];
    const float* nb1     = (const float*)d_in[18];
    const float* ng1     = (const float*)d_in[19];
    const float* nbe1    = (const float*)d_in[20];
    const float* nw2     = (const float*)d_in[21];
    const float* nb2     = (const float*)d_in[22];
    const float* key_w   = (const float*)d_in[23];
    const float* key_b   = (const float*)d_in[24];
    const float* val_w   = (const float*)d_in[25];
    const float* val_b   = (const float*)d_in[26];
    // d_in[27]=q_w, d_in[28]=q_b: dead code (softmax over a singleton axis == 1)
    const float* rd_w1   = (const float*)d_in[29];
    const float* rd_b1   = (const float*)d_in[30];
    const float* rd_g1   = (const float*)d_in[31];
    const float* rd_be1  = (const float*)d_in[32];
    const float* rd_w2   = (const float*)d_in[33];
    const float* rd_b2   = (const float*)d_in[34];
    const float* rd_g2   = (const float*)d_in[35];
    const float* rd_be2  = (const float*)d_in[36];
    const float* rd_w3   = (const float*)d_in[37];
    const float* rd_b3   = (const float*)d_in[38];
    const float* init_pre  = (const float*)d_in[39];
    const float* init_post = (const float*)d_in[40];
    const float* decay   = (const float*)d_in[41];
    const int*   pairs   = (const int*)d_in[42];

    float* ws = (float*)d_ws;
    size_t off = 0;
    auto carve   = [&](size_t n) { float* p = ws + off; off += n; return p; };
    auto carve_u = [&](size_t n) { unsigned int* p = (unsigned int*)(ws + off); off += n; return p; };
    auto swsz    = [&](int Kd, int Nd) { return (size_t)cdiv(Kd, 32) * cdiv(Nd, 64) * 1024; };

    float* pre      = carve((size_t)Bb * Nn * Mm);
    float* post     = carve((size_t)Bb * Nn * Mm);
    float* h1       = carve((size_t)Bb * 128);
    float* h2       = carve((size_t)Bb * 32);
    float* enc      = carve((size_t)Bb * Rr);
    float* keys     = carve((size_t)Bb * Aa);
    float* values   = carve((size_t)Bb * Aa);
    float* syn_in   = carve((size_t)Bb * (Rr + Nn));
    float* sh       = carve((size_t)Bb * 256);
    float* new_pre  = carve((size_t)Bb * Nn);
    float* post_new = carve((size_t)Bb * Nn);
    float* syncv    = carve((size_t)Bb * Pp);
    float* comb     = carve((size_t)Bb * (Aa + Pp));
    float* rh1      = carve((size_t)Bb * 512);
    float* rh2      = carve((size_t)Bb * 64);
    float* pred     = carve((size_t)Bb * 10);
    int*   done     = (int*)carve(4);
    unsigned int* sw_ie1 = carve_u(swsz(784, 128));
    unsigned int* sw_ie2 = carve_u(swsz(128, 32));
    unsigned int* sw_ie3 = carve_u(swsz(32, Rr));
    unsigned int* sw_sy1 = carve_u(swsz(Rr + Nn, 256));
    unsigned int* sw_sy2 = carve_u(swsz(256, Nn));
    unsigned int* sw_rd1 = carve_u(swsz(Aa + Pp, 512));
    unsigned int* sw_rd2 = carve_u(swsz(512, 64));
    unsigned int* sw_rd3 = carve_u(swsz(64, 10));
    float* stored   = (float*)d_out;

    auto swz = [&](const float* W, unsigned int* out_, int Kd, int Nd) {
        const int Kt = cdiv(Kd, 32), Nt = cdiv(Nd, 64);
        const int tot = Kt * Nt * 1024;
        swizzle_weight<<<cdiv(tot, 256), 256, 0, stream>>>(W, out_, Kd, Nd, Kt, Nt);
    };
    auto gemm = [&](const float* A, const unsigned int* Bsw, const float* bias,
                    float* C, int Kd, int Nd) {
        const int Nt = cdiv(Nd, 64);
        gemm64_wmma<<<Nt, 128, 0, stream>>>(A, Bsw, bias, C, Kd, Nd, Nt);
    };
    auto lnrelu = [&](float* buf, const float* g_, const float* b_, int rows, int D) {
        ln_relu_rows<<<rows, 32, 0, stream>>>(buf, g_, b_, rows, D);
    };

    // ---- once: pre-swizzle weights to fragment layout (bf16), init state ----
    swz(ie_w1, sw_ie1, 784, 128);
    swz(ie_w2, sw_ie2, 128, 32);
    swz(ie_w3, sw_ie3, 32, Rr);
    swz(sy_w1, sw_sy1, Rr + Nn, 256);
    swz(sy_w2, sw_sy2, 256, Nn);
    swz(rd_w1, sw_rd1, Aa + Pp, 512);
    swz(rd_w2, sw_rd2, 512, 64);
    swz(rd_w3, sw_rd3, 64, 10);
    {
        const size_t tot = (size_t)Bb * Nn * Mm;
        broadcast_init<<<cdiv((int)tot, 256), 256, 0, stream>>>(init_pre, pre);
        broadcast_init<<<cdiv((int)tot, 256), 256, 0, stream>>>(init_post, post);
        zero_int<<<1, 1, 0, stream>>>(done);
    }

    // ---- encoder: x(64,784) -> 128 -> 32 -> enc(64,4); keys/values ----
    gemm(x, sw_ie1, ie_b1, h1, 784, 128);
    lnrelu(h1, ie_g1, ie_be1, Bb, 128);
    gemm(h1, sw_ie2, ie_b2, h2, 128, 32);
    lnrelu(h2, ie_g2, ie_be2, Bb, 32);
    gemm(h2, sw_ie3, ie_b3, enc, 32, Rr);
    kv_kernel<<<cdiv(Bb * Aa, 256), 256, 0, stream>>>(enc, key_w, key_b, val_w, val_b,
                                                      keys, values);

    // ---- T sequential ticks ----
    for (int t = 0; t < Tt; ++t) {
        build_syn_in<<<cdiv(Bb * (Rr + Nn), 256), 256, 0, stream>>>(enc, post, syn_in);

        gemm(syn_in, sw_sy1, sy_b1, sh, Rr + Nn, 256);        // (64x4100)@(4100x256)
        lnrelu(sh, sy_g1, sy_be1, Bb, 256);
        gemm(sh, sw_sy2, sy_b2, new_pre, 256, Nn);            // (64x256)@(256x4096)

        shift_append<<<cdiv(Bb * Nn, 256), 256, 0, stream>>>(pre, new_pre);

        neuron_tick<<<Nn, 256, 0, stream>>>(pre, nw1, nb1, ng1, nbe1, nw2, nb2,
                                            post_new);

        shift_append<<<cdiv(Bb * Nn, 256), 256, 0, stream>>>(post, post_new);

        sync_kernel<<<cdiv(Bb * Pp, 256), 256, 0, stream>>>(post, pairs, decay, t, syncv);
        build_comb<<<cdiv(Bb * (Aa + Pp), 256), 256, 0, stream>>>(values, syncv, comb);

        gemm(comb, sw_rd1, rd_b1, rh1, Aa + Pp, 512);         // (64x1088)@(1088x512)
        lnrelu(rh1, rd_g1, rd_be1, Bb, 512);
        gemm(rh1, sw_rd2, rd_b2, rh2, 512, 64);
        lnrelu(rh2, rd_g2, rd_be2, Bb, 64);
        gemm(rh2, sw_rd3, rd_b3, pred, 64, 10);

        finalize_kernel<<<1, Bb, 0, stream>>>(pred, stored, done);
    }
}
